// CrossExpertRefinement_12893491823346
// MI455X (gfx1250) — matrix operand order repaired and checked by reading, not compiled, not run on hardware
//
#include <hip/hip_runtime.h>
#include <hip/hip_bf16.h>

typedef __attribute__((ext_vector_type(16))) __bf16       v16bf_t;
typedef __attribute__((ext_vector_type(8)))  float        v8f_t;
typedef __attribute__((ext_vector_type(4)))  unsigned int u32x4_t;
typedef __attribute__((ext_vector_type(2)))  float        f32x2_t;
typedef __attribute__((ext_vector_type(4)))  float        f32x4_t;

#define TOK_TILE 16
#define ROWS 64          // TOK_TILE tokens * 4 experts
#define NTHREADS 256
#define NWAVES 8

// ---------------------------------------------------------------------------
// Explicit global-address-space accessors (avoid flat_* codegen).
// ---------------------------------------------------------------------------
typedef __attribute__((address_space(1))) const float*    gc_f32;
typedef __attribute__((address_space(1)))       float*    g_f32;
typedef __attribute__((address_space(1))) const u32x4_t*  gc_u4;
typedef __attribute__((address_space(1))) const f32x2_t*  gc_f2;
typedef __attribute__((address_space(1))) const f32x4_t*  gc_f4;
typedef __attribute__((address_space(1)))       __bf16*   g_bf16;

__device__ inline float gload_f32(const float* p) {
  return *(gc_f32)(unsigned long long)p;
}
__device__ inline u32x4_t gload_u4(const void* p) {
  return *(gc_u4)(unsigned long long)p;
}
__device__ inline f32x2_t gload_f2(const float* p) {
  return *(gc_f2)(unsigned long long)p;
}
__device__ inline f32x4_t gload_f4(const float* p) {
  return *(gc_f4)(unsigned long long)p;
}
__device__ inline void gstore_f32(float* p, float v) {
  *(g_f32)(unsigned long long)p = v;
}
__device__ inline void gstore_bf16(__bf16* p, __bf16 v) {
  *(g_bf16)(unsigned long long)p = v;
}

__device__ inline unsigned pack_bf16x2(float a, float b) {
  union { __bf16 h[2]; unsigned u; } p;
  p.h[0] = (__bf16)a;
  p.h[1] = (__bf16)b;
  return p.u;
}

// ---------------------------------------------------------------------------
// WMMA wrapper: D = A(16x32 bf16) * B(32x16 bf16) + C(16x16 f32)
// ---------------------------------------------------------------------------
__device__ inline v8f_t wmma_bf16(v16bf_t a, v16bf_t b, v8f_t c) {
  return __builtin_amdgcn_wmma_f32_16x16x32_bf16(
      false, a, false, b, (short)0, c, false, false);
}

union FragBF {
  v16bf_t v;
  __bf16 h[16];
  u32x4_t q[2];
};

// A fragment (16x32, lane = M, halves: K = kb+8*lo+t and kb+16+8*lo+t)
__device__ inline v16bf_t load_a_frag_lds(const __bf16* buf, int row, int ld,
                                          int kb, int lo) {
  FragBF u;
  const __bf16* p = buf + row * ld + kb + lo * 8;
  u.q[0] = ((const u32x4_t*)p)[0];
  u.q[1] = ((const u32x4_t*)(p + 16))[0];
  return u.v;
}

// B fragment (32x16, lane = N mod 16, halves: K = kb+8*lo+t and kb+16+8*lo+t)
__device__ inline v16bf_t load_b_frag_g(const __bf16* base, int n, int ldk,
                                        int kb, int lo) {
  FragBF u;
  const __bf16* p = base + (size_t)n * ldk + kb + 8 * lo;
  u.q[0] = gload_u4(p);
  u.q[1] = gload_u4(p + 16);
  return u.v;
}

// ---------------------------------------------------------------------------
// Weight convert fp32 -> bf16 with row (R->Rpad) and column (C->Cpad) zero pad.
// ---------------------------------------------------------------------------
__global__ void cvt_pad_kernel(const float* __restrict__ src, __bf16* __restrict__ dst,
                               int R, int C, int Rpad, int Cpad) {
  long long idx = (long long)blockIdx.x * blockDim.x + threadIdx.x;
  long long total = (long long)Rpad * Cpad;
  if (idx >= total) return;
  int r = (int)(idx / Cpad);
  int c = (int)(idx % Cpad);
  float v = (r < R && c < C) ? gload_f32(src + (long long)r * C + c) : 0.f;
  gstore_bf16(dst + idx, (__bf16)v);
}

// ---------------------------------------------------------------------------
// Stage 1 (one expert, compile-time dims): stage x tile (float2, packed bf16)
// into LDS, then stacked[:, e, :] = x_e @ Win_e^T.  M = 16 tokens, fully used.
// ---------------------------------------------------------------------------
template <int D, int DPAD, int SHIFTC>
__device__ inline void stage1_expert(const float* __restrict__ xg,
                                     const __bf16* __restrict__ WinBe,
                                     __bf16* sX, float* sStacked,
                                     int e, int tokBase,
                                     int tid, int wave, int lane, int lo, int mrow) {
  constexpr int CH = D / 2;        // real float2 chunks per row (D is even)
  constexpr int CHP = DPAD / 2;    // padded chunks per row (power of two)
  unsigned* sX32 = (unsigned*)sX;
#pragma unroll 2
  for (int idx = tid; idx < TOK_TILE * CHP; idx += NTHREADS) {
    const int row = idx >> SHIFTC;
    const int c2 = idx & (CHP - 1);
    float a = 0.f, b = 0.f;
    if (c2 < CH) {
      f32x2_t w = gload_f2(xg + (size_t)(tokBase + row) * D + 2 * c2);
      a = w.x; b = w.y;
    }
    sX32[idx] = pack_bf16x2(a, b);
  }
  __syncthreads();
  for (int ntile = wave; ntile < 32; ntile += NWAVES) {
    v8f_t c;
#pragma unroll
    for (int v = 0; v < 8; ++v) c[v] = 0.f;
#pragma unroll 4
    for (int ks = 0; ks < DPAD / 32; ++ks) {
      const int kb = ks << 5;
      v16bf_t a = load_a_frag_lds(sX, mrow, DPAD, kb, lo);
      v16bf_t b = load_b_frag_g(WinBe, ntile * 16 + mrow, DPAD, kb, lo);
      c = wmma_bf16(a, b, c);
    }
#pragma unroll
    for (int v = 0; v < 8; ++v) {      // M = v + 8*lo = token-in-tile
      const int M = v + 8 * lo;
      sStacked[(M * 4 + e) * 512 + ntile * 16 + mrow] = c[v];
    }
  }
  __syncthreads();
}

// ---------------------------------------------------------------------------
// Fully fused cross-expert refinement: one block = 16 tokens, 8 waves.
// ---------------------------------------------------------------------------
__global__ __launch_bounds__(NTHREADS) void fused_cross_expert(
    const float* __restrict__ x0, const float* __restrict__ x1,
    const float* __restrict__ x2, const float* __restrict__ x3,
    const float* __restrict__ ln_g, const float* __restrict__ ln_b,
    const float* __restrict__ bqkv, const float* __restrict__ bo,
    const __bf16* __restrict__ wt, float* __restrict__ out) {
  __shared__ __align__(16) float  sStacked[ROWS * 512];  // 128 KB fp32 residual
  __shared__ __align__(16) __bf16 sNormed[ROWS * 512];   // 64 KB (normed, then stacked-bf16)
  __shared__ __align__(16) float  sQKV[ROWS * 192];      // 48 KB per-head q|k|v chunk
  __shared__ __align__(16) __bf16 sCtx[ROWS * 512];      // 64 KB (x staging overlay in stage 1)
  __shared__ float sRedS[ROWS * 4];
  __shared__ float sRedS2[ROWS * 4];
  __shared__ float sMu[ROWS];
  __shared__ float sRs[ROWS];

  const int tid  = threadIdx.x;
  const int wave = tid >> 5;
  const int lane = tid & 31;
  const int lo   = lane >> 4;
  const int mrow = lane & 15;
  const int tokBase = blockIdx.x * TOK_TILE;

  // ---------------- Stage 1: stacked = x_e @ Win_e^T ----------------------
  {
    __bf16* sX = sCtx;  // overlay: sCtx unused until stage 3 (16*1024 <= 64*512)
    stage1_expert<128, 128, 6>(x0, wt + 0,        sX, sStacked, 0, tokBase, tid, wave, lane, lo, mrow);
    stage1_expert<128, 128, 6>(x1, wt + 65536,    sX, sStacked, 1, tokBase, tid, wave, lane, lo, mrow);
    stage1_expert<1006, 1024, 9>(x2, wt + 131072, sX, sStacked, 2, tokBase, tid, wave, lane, lo, mrow);
    stage1_expert<104, 128, 6>(x3, wt + 655360,   sX, sStacked, 3, tokBase, tid, wave, lane, lo, mrow);
  }

  // ---------------- Stage 2: LayerNorm -> sNormed (bf16) ------------------
  {
    const int r = tid >> 2;          // 0..63 (256 threads / 4 per row)
    const int qd = tid & 3;
    const f32x4_t* row4 = (const f32x4_t*)(sStacked + r * 512);
    float s = 0.f, s2 = 0.f;
#pragma unroll 8
    for (int t = 0; t < 32; ++t) {
      f32x4_t v = row4[qd * 32 + t];
      s += v.x + v.y + v.z + v.w;
      s2 += v.x * v.x + v.y * v.y + v.z * v.z + v.w * v.w;
    }
    sRedS[r * 4 + qd] = s;
    sRedS2[r * 4 + qd] = s2;
    __syncthreads();
    if (qd == 0) {
      float S = 0.f, S2 = 0.f;
#pragma unroll
      for (int i = 0; i < 4; ++i) { S += sRedS[r * 4 + i]; S2 += sRedS2[r * 4 + i]; }
      float mu = S * (1.f / 512.f);
      float var = S2 * (1.f / 512.f) - mu * mu;
      sMu[r] = mu;
      sRs[r] = rsqrtf(var + 1e-5f);
    }
    __syncthreads();
    const float mu = sMu[r];
    const float rs = sRs[r];
    unsigned* n32 = (unsigned*)sNormed;
#pragma unroll 4
    for (int t = 0; t < 32; ++t) {
      const int cc = qd * 128 + t * 4;
      f32x4_t v = row4[qd * 32 + t];
      f32x4_t g = gload_f4(ln_g + cc);
      f32x4_t b = gload_f4(ln_b + cc);
      float n0 = (v.x - mu) * rs * g.x + b.x;
      float n1 = (v.y - mu) * rs * g.y + b.y;
      float n2 = (v.z - mu) * rs * g.z + b.z;
      float n3 = (v.w - mu) * rs * g.w + b.w;
      const int base = (r * 512 + cc) >> 1;
      n32[base]     = pack_bf16x2(n0, n1);
      n32[base + 1] = pack_bf16x2(n2, n3);
    }
  }
  __syncthreads();

  // ---------------- Stage 3: per-head QKV GEMM + masked attention ---------
  const __bf16* WqkvB = wt + 720896;   // 1536 x 512
  for (int h = 0; h < 8; ++h) {
    // 24 jobs = 2 mtile-pairs x 12 ntiles; one B fragment feeds two WMMAs.
    for (int job = wave; job < 24; job += NWAVES) {
      const int mp  = job / 12;
      const int nt  = job % 12;
      const int g   = nt >> 2;           // 0=q 1=k 2=v
      const int sub = nt & 3;
      const int nglob = g * 512 + h * 64 + sub * 16;
      const int lcol  = g * 64 + sub * 16;
      const int m0 = mp * 32, m1 = mp * 32 + 16;
      v8f_t c0, c1;
      const float bias = gload_f32(bqkv + nglob + mrow);
#pragma unroll
      for (int v = 0; v < 8; ++v) { c0[v] = bias; c1[v] = bias; }
#pragma unroll 4
      for (int ks = 0; ks < 16; ++ks) {
        const int kb = ks << 5;
        v16bf_t b  = load_b_frag_g(WqkvB, nglob + mrow, 512, kb, lo);
        v16bf_t a0 = load_a_frag_lds(sNormed, m0 + mrow, 512, kb, lo);
        v16bf_t a1 = load_a_frag_lds(sNormed, m1 + mrow, 512, kb, lo);
        c0 = wmma_bf16(a0, b, c0);
        c1 = wmma_bf16(a1, b, c1);
      }
#pragma unroll
      for (int v = 0; v < 8; ++v) {
        sQKV[(m0 + v + 8 * lo) * 192 + lcol + mrow] = c0[v];
        sQKV[(m1 + v + 8 * lo) * 192 + lcol + mrow] = c1[v];
      }
    }
    __syncthreads();

    {   // attention: 256 threads = 16 tokens x 4 rows x 4 column-quarters
      const int tok = tid >> 4;
      const int i   = (tid >> 2) & 3;
      const int cq  = tid & 3;
      const int rb  = tok * 4;
      const f32x4_t* q4 = (const f32x4_t*)(sQKV + (rb + i) * 192);
      float sc[4];
      float mx = -1e30f;
#pragma unroll
      for (int j = 0; j < 4; ++j) {
        if (j <= i) {
          const f32x4_t* k4 = (const f32x4_t*)(sQKV + (rb + j) * 192 + 64);
          float d = 0.f;
#pragma unroll
          for (int t = 0; t < 16; ++t) {
            f32x4_t qa = q4[t], kk = k4[t];
            d += qa.x * kk.x + qa.y * kk.y + qa.z * kk.z + qa.w * kk.w;
          }
          sc[j] = d * 0.125f;
          mx = fmaxf(mx, sc[j]);
        }
      }
      float den = 0.f;
#pragma unroll
      for (int j = 0; j < 4; ++j) {
        if (j <= i) { sc[j] = __expf(sc[j] - mx); den += sc[j]; }
        else sc[j] = 0.f;
      }
      const float inv = 1.f / den;
      unsigned* c32 = (unsigned*)sCtx;
#pragma unroll
      for (int t = 0; t < 4; ++t) {
        const int cc = cq * 16 + t * 4;
        f32x4_t acc = {0.f, 0.f, 0.f, 0.f};
#pragma unroll
        for (int j = 0; j < 4; ++j) {
          if (j <= i) {
            const f32x4_t* v4 = (const f32x4_t*)(sQKV + (rb + j) * 192 + 128);
            acc += sc[j] * v4[cc >> 2];
          }
        }
        const int base = ((rb + i) * 512 + h * 64 + cc) >> 1;
        c32[base]     = pack_bf16x2(acc.x * inv, acc.y * inv);
        c32[base + 1] = pack_bf16x2(acc.z * inv, acc.w * inv);
      }
    }
    __syncthreads();
  }

  // ---------------- Stage 4: stacked += ctx @ Wo^T + bo -------------------
  const __bf16* WoB = wt + 1507328;    // 512 x 512
  for (int job = wave; job < 64; job += NWAVES) {  // 2 mtile-pairs x 32 ntiles
    const int mp = job >> 5;
    const int nb = (job & 31) * 16;
    const int m0 = mp * 32, m1 = mp * 32 + 16;
    v8f_t c0, c1;
    const float bias = gload_f32(bo + nb + mrow);
#pragma unroll
    for (int v = 0; v < 8; ++v) { c0[v] = bias; c1[v] = bias; }
#pragma unroll 4
    for (int ks = 0; ks < 16; ++ks) {
      const int kb = ks << 5;
      v16bf_t b  = load_b_frag_g(WoB, nb + mrow, 512, kb, lo);
      v16bf_t a0 = load_a_frag_lds(sCtx, m0 + mrow, 512, kb, lo);
      v16bf_t a1 = load_a_frag_lds(sCtx, m1 + mrow, 512, kb, lo);
      c0 = wmma_bf16(a0, b, c0);
      c1 = wmma_bf16(a1, b, c1);
    }
#pragma unroll
    for (int v = 0; v < 8; ++v) {
      sStacked[(m0 + v + 8 * lo) * 512 + nb + mrow] += c0[v];
      sStacked[(m1 + v + 8 * lo) * 512 + nb + mrow] += c1[v];
    }
  }
  __syncthreads();

  // ---------------- Stage 5 prep: stacked -> bf16 (reuse sNormed) ---------
  {
    const f32x2_t* st2 = (const f32x2_t*)sStacked;
    unsigned* n32 = (unsigned*)sNormed;
#pragma unroll 4
    for (int idx = tid; idx < ROWS * 256; idx += NTHREADS) {
      f32x2_t v = st2[idx];
      n32[idx] = pack_bf16x2(v.x, v.y);
    }
  }
  __syncthreads();

  // ---------------- Stage 5: out_e = stacked_e @ Wout_e^T + x_e -----------
  for (int job = wave; job < 86; job += NWAVES) {
    int e, nt, d;
    size_t oOff;
    const __bf16* Bw;
    const float* xsrc;
    if (job < 8)       { e = 0; nt = job;      d = 128;  oOff = 0u;         Bw = wt + 1769472; xsrc = x0; }
    else if (job < 16) { e = 1; nt = job - 8;  d = 128;  oOff = 2097152u;   Bw = wt + 1835008; xsrc = x1; }
    else if (job < 79) { e = 2; nt = job - 16; d = 1006; oOff = 4194304u;   Bw = wt + 1900544; xsrc = x2; }
    else               { e = 3; nt = job - 79; d = 104;  oOff = 20676608u;  Bw = wt + 2416640; xsrc = x3; }
    const int arow = mrow * 4 + e;   // token row in sNormed (all 16 valid)
    v8f_t c;
#pragma unroll
    for (int v = 0; v < 8; ++v) c[v] = 0.f;
#pragma unroll 4
    for (int ks = 0; ks < 16; ++ks) {
      const int kb = ks << 5;
      v16bf_t a = load_a_frag_lds(sNormed, arow, 512, kb, lo);
      v16bf_t b = load_b_frag_g(Bw, nt * 16 + mrow, 512, kb, lo);  // N row-padded
      c = wmma_bf16(a, b, c);
    }
    const int n = nt * 16 + mrow;
    if (n < d) {
      float* op = out + oOff;
#pragma unroll
      for (int v = 0; v < 8; ++v) {          // M = v + 8*lo = token-in-tile
        const size_t tok = (size_t)(tokBase + v + 8 * lo);
        gstore_f32(op + tok * d + n, c[v] + gload_f32(xsrc + tok * d + n));
      }
    }
  }
}

// ---------------------------------------------------------------------------
extern "C" void kernel_launch(void* const* d_in, const int* in_sizes, int n_in,
                              void* d_out, int out_size, void* d_ws, size_t ws_size,
                              hipStream_t stream) {
  (void)in_sizes; (void)n_in; (void)out_size; (void)ws_size;
  const float* x0    = (const float*)d_in[0];
  const float* Win0  = (const float*)d_in[1];
  const float* Wout0 = (const float*)d_in[2];
  const float* x1    = (const float*)d_in[3];
  const float* Win1  = (const float*)d_in[4];
  const float* Wout1 = (const float*)d_in[5];
  const float* x2    = (const float*)d_in[6];
  const float* Win2  = (const float*)d_in[7];
  const float* Wout2 = (const float*)d_in[8];
  const float* x3    = (const float*)d_in[9];
  const float* Win3  = (const float*)d_in[10];
  const float* Wout3 = (const float*)d_in[11];
  const float* ln_g  = (const float*)d_in[12];
  const float* ln_b  = (const float*)d_in[13];
  const float* Wqkv  = (const float*)d_in[14];
  const float* bqkv  = (const float*)d_in[15];
  const float* Wo    = (const float*)d_in[16];
  const float* bo    = (const float*)d_in[17];

  __bf16* wt = (__bf16*)d_ws;

  auto launch_cvt = [&](const float* src, size_t off, int R, int C, int Rpad, int Cpad) {
    long long total = (long long)Rpad * Cpad;
    int blocks = (int)((total + 255) / 256);
    cvt_pad_kernel<<<blocks, 256, 0, stream>>>(src, wt + off, R, C, Rpad, Cpad);
  };
  // Win_e: (512 x d_e), K padded to mult-of-32
  launch_cvt(Win0, 0u,       512, 128,  512, 128);
  launch_cvt(Win1, 65536u,   512, 128,  512, 128);
  launch_cvt(Win2, 131072u,  512, 1006, 512, 1024);
  launch_cvt(Win3, 655360u,  512, 104,  512, 128);
  // Wqkv (1536 x 512), Wo (512 x 512)
  launch_cvt(Wqkv, 720896u,  1536, 512, 1536, 512);
  launch_cvt(Wo,   1507328u, 512, 512,  512, 512);
  // Wout_e: (d_e x 512), N (rows) padded to mult-of-16
  launch_cvt(Wout0, 1769472u, 128, 512,  128, 512);
  launch_cvt(Wout1, 1835008u, 128, 512,  128, 512);
  launch_cvt(Wout2, 1900544u, 1006, 512, 1008, 512);
  launch_cvt(Wout3, 2416640u, 104, 512,  112, 512);

  const int nTokens = 8 * 2048;             // B*T = 16384
  const int nBlocks = nTokens / TOK_TILE;   // 1024
  fused_cross_expert<<<nBlocks, NTHREADS, 0, stream>>>(
      x0, x1, x2, x3, ln_g, ln_b, bqkv, bo, wt, (float*)d_out);
}
